// MOE_90735479095717
// MI455X (gfx1250) — matrix operand order repaired and checked by reading, not compiled, not run on hardware
//
#include <hip/hip_runtime.h>
#include <hip/hip_bf16.h>

// ---------------------------------------------------------------------------
// MoE (top-2 of 8 experts), H=1024, I=2816, N=4096 tokens, fp32 in/out.
// gate -> route per expert -> grouped SwiGLU GEMMs on V_WMMA_F32_16X16X32_BF16
// with 3-term bf16 hi/lo split (near-fp32 accuracy, 2.7x fewer matrix instrs
// than native-f32 WMMA) -> weighted f32 atomic combine.
// Per-expert weights (34.6MB) are re-read from the 192MB L2 across token
// tiles; HBM floor is one pass over 277MB of weights (~12us @ 23.3TB/s).
// Software pipeline per K-step: convert(prev loads) -> store buf[next] ->
// issue loads for step+2 -> WMMA from buf[cur]. Loads are consumed at the
// TOP of the following iteration, so no v_mov pipeline copies and no
// s_wait_loadcnt in front of the matrix ops.
// ---------------------------------------------------------------------------

#define HDIM   1024
#define IDIM   2816
#define NEXP   8
#define NTOK   4096            // B*S = 2*2048
#define TW     16              // tokens per tile (one WMMA M)
#define KB_A   (HDIM / 32)     // 32 k-steps for gate/up proj
#define IT_A   (IDIM / 128)    // 22 column chunks of 128 (8 waves x 16 cols)
#define KB_B   (IDIM / 32)     // 88 k-steps for down proj
// dynamic LDS: Hm hi/lo (2*16*2816 us) + union region (65536 us)
#define SMEM_US (2 * TW * IDIM + 65536)
#define SMEM_BYTES (SMEM_US * 2)

typedef __bf16 v16bf __attribute__((ext_vector_type(16)));
typedef float  v8f   __attribute__((ext_vector_type(8)));

__device__ __forceinline__ unsigned short f2bf_rn(float x) {
    unsigned u = __float_as_uint(x);
    unsigned r = u + 0x7FFFu + ((u >> 16) & 1u);
    return (unsigned short)(r >> 16);
}
__device__ __forceinline__ float bf2f(unsigned short b) {
    return __uint_as_float(((unsigned)b) << 16);
}
__device__ __forceinline__ void split_store(float x, unsigned short* hi,
                                            unsigned short* lo, int off) {
    unsigned short h = f2bf_rn(x);
    hi[off] = h;
    lo[off] = f2bf_rn(x - bf2f(h));
}
// pack a column of 4 consecutive K-rows into one b64 store per hi/lo array
__device__ __forceinline__ void split4_store(const float* v, unsigned short* hi,
                                             unsigned short* lo, int off) {
    unsigned short h[4], l[4];
#pragma unroll
    for (int i = 0; i < 4; ++i) {
        h[i] = f2bf_rn(v[i]);
        l[i] = f2bf_rn(v[i] - bf2f(h[i]));
    }
    uint2 ph, pl;
    ph.x = (unsigned)h[0] | ((unsigned)h[1] << 16);
    ph.y = (unsigned)h[2] | ((unsigned)h[3] << 16);
    pl.x = (unsigned)l[0] | ((unsigned)l[1] << 16);
    pl.y = (unsigned)l[2] | ((unsigned)l[3] << 16);
    *(uint2*)(hi + off) = ph;   // ds_store_b64
    *(uint2*)(lo + off) = pl;   // ds_store_b64
}
// convert a row-quad of float4 columns into hi/lo B-fragments
__device__ __forceinline__ void cvt_store_quad(const float4& q0, const float4& q1,
                                               const float4& q2, const float4& q3,
                                               unsigned short* hi, unsigned short* lo,
                                               int c4, int half, int j0) {
#pragma unroll
    for (int z = 0; z < 4; ++z) {
        int c = c4 + z;
        int off = ((c >> 4) * 32 + (c & 15) + half) * 16 + j0;
        float t[4] = {((const float*)&q0)[z], ((const float*)&q1)[z],
                      ((const float*)&q2)[z], ((const float*)&q3)[z]};
        split4_store(t, hi, lo, off);
    }
}
// A-matrix (16x32 bf16) fragment mapping: row t, k-rel kr -> (lane, elem j)
__device__ __forceinline__ int a_lane_j(int t, int kr, int& lane2) {
    lane2 = t + ((kr >> 4) << 4);
    return kr & 15;
}
__device__ __forceinline__ v8f wmma_bf16(v16bf a, v16bf b, v8f c) {
    return __builtin_amdgcn_wmma_f32_16x16x32_bf16(
        false, a, false, b, (short)0, c, false, false);
}

// ---------------------------------------------------------------------------
__global__ void moe_init_kernel(int* cnt, int* idxbuf) {
    int i = blockIdx.x * blockDim.x + threadIdx.x;
    if (i < NEXP * NTOK) idxbuf[i] = -1;
    if (i < NEXP) cnt[i] = 0;
}

__global__ void moe_zero_out_kernel(float* out) {
    size_t i = (size_t)blockIdx.x * blockDim.x + threadIdx.x;
    if (i < (size_t)NTOK * HDIM) out[i] = 0.0f;
}

// one wave32 per token: logits = x @ Wgate + bgate, pick top-2
__global__ void moe_gate_kernel(const float* __restrict__ x,
                                const float* __restrict__ Wgate,
                                const float* __restrict__ bgate,
                                float* __restrict__ logits_out,
                                int* __restrict__ cnt,
                                int* __restrict__ idxbuf,
                                float* __restrict__ wbuf) {
    int gtid = blockIdx.x * blockDim.x + threadIdx.x;
    int tok  = gtid >> 5;
    int lane = gtid & 31;
    if (tok >= NTOK) return;

    float acc[NEXP];
#pragma unroll
    for (int e = 0; e < NEXP; ++e) acc[e] = 0.0f;

    const float* xr = x + (size_t)tok * HDIM;
    for (int h = lane; h < HDIM; h += 32) {
        float xv = xr[h];
        const float* wr = Wgate + (size_t)h * NEXP;
#pragma unroll
        for (int e = 0; e < NEXP; ++e) acc[e] += xv * wr[e];
    }
#pragma unroll
    for (int off = 16; off > 0; off >>= 1) {
#pragma unroll
        for (int e = 0; e < NEXP; ++e) acc[e] += __shfl_xor(acc[e], off);
    }
    if (lane == 0) {
#pragma unroll
        for (int e = 0; e < NEXP; ++e) {
            acc[e] += bgate[e];
            logits_out[(size_t)tok * NEXP + e] = acc[e];
        }
        // top-2 (ties -> lower index, matching lax.top_k)
        int i0 = 0; float v0 = acc[0];
#pragma unroll
        for (int e = 1; e < NEXP; ++e)
            if (acc[e] > v0) { v0 = acc[e]; i0 = e; }
        int i1 = -1; float v1 = -3.4e38f;
#pragma unroll
        for (int e = 0; e < NEXP; ++e)
            if (e != i0 && acc[e] > v1) { v1 = acc[e]; i1 = e; }
        int p0 = atomicAdd(&cnt[i0], 1);
        idxbuf[i0 * NTOK + p0] = tok;
        wbuf[i0 * NTOK + p0]   = v0;
        int p1 = atomicAdd(&cnt[i1], 1);
        idxbuf[i1 * NTOK + p1] = tok;
        wbuf[i1 * NTOK + p1]   = v1;
    }
}

// ---------------------------------------------------------------------------
// Fused per-expert SwiGLU: block = (expert, 16-token tile), 8 waves.
// Phase A: G,U = X@Wg, X@Wu  (split-bf16 WMMA), Hm = silu(G)*U -> LDS.
// Phase B: O = Hm@Wd, scaled atomic add into out. Both phases pipelined.
__global__ void __launch_bounds__(256)
moe_expert_kernel(const float* __restrict__ x,
                  const float* __restrict__ Wg,
                  const float* __restrict__ Wu,
                  const float* __restrict__ Wd,
                  const int* __restrict__ cnt,
                  const int* __restrict__ idxbuf,
                  const float* __restrict__ wbuf,
                  float* __restrict__ out) {
    extern __shared__ unsigned short smem[];
    __shared__ int   toks[TW];
    __shared__ float tws[TW];

    const int e    = blockIdx.y;
    const int nt   = cnt[e];
    const int row0 = blockIdx.x * TW;
    if (row0 >= nt) return;

    unsigned short* HmHi = smem;                    // 16*2816
    unsigned short* HmLo = HmHi + TW * IDIM;        // 16*2816
    unsigned short* R    = HmLo + TW * IDIM;        // union region (65536 us)
    // phase A view: A-tile + double-buffered Wg/Wu fragment staging
    unsigned short* AHi  = R;                       // 16*1024
    unsigned short* ALo  = AHi + TW * HDIM;
    unsigned short* SB   = ALo + TW * HDIM;         // 2 x 16384 us buffers
                                                    // [gHi|gLo|uHi|uLo] x 4096
    // phase B view: double-buffered Wd quarter staging (2 x 16384 us)
    // buffer b: [hi 8192 | lo 8192] at R + b*16384

    const int tid  = threadIdx.x;
    const int wave = tid >> 5;
    const int lane = tid & 31;

    if (tid < TW) {
        int slot = row0 + tid;
        int t = idxbuf[e * NTOK + slot];
        toks[tid] = t;
        tws[tid]  = (t >= 0) ? wbuf[e * NTOK + slot] : 0.0f;
    }
    __syncthreads();

    // stage X tile [16 x 1024] -> bf16 hi/lo A-fragments, quad-packed b64
#pragma unroll 4
    for (int q = 0; q < 16; ++q) {
        int li = tid + q * 256;            // 4096 h-quads
        int r  = li >> 8;                  // token row
        int h0 = (li & 255) * 4;           // 4-aligned h
        int t  = toks[r];
        float4 xv = make_float4(0.f, 0.f, 0.f, 0.f);
        if (t >= 0) xv = *(const float4*)&x[(size_t)t * HDIM + h0];
        int kb = h0 >> 5, kr0 = h0 & 31;
        int lane2; int j0 = a_lane_j(r, kr0, lane2);
        float tv[4] = {xv.x, xv.y, xv.z, xv.w};
        split4_store(tv, AHi, ALo, (kb * 32 + lane2) * 16 + j0);
    }
    __syncthreads();

    // -------- Phase A: G/U over 128-column chunks, pipelined staging ------
    const int rqA   = tid >> 5;            // row quad 0..7
    const int kr0A  = rqA * 4;
    const int c4A   = (tid & 31) * 4;
    const int halfA = (kr0A >= 16) ? 16 : 0;
    const int j0A   = kr0A & 15;

    for (int it = 0; it < IT_A; ++it) {
        const int colbase = it * 128;
        v8f cg = {}; v8f cu = {};
        float4 g0, g1, g2, g3, u0, u1, u2, u3;
        const size_t base0 = ((size_t)e * HDIM + kr0A) * IDIM + colbase + c4A;
        {   // prologue: stage step 0 into buffer 0, then issue step-1 loads
            g0 = *(const float4*)&Wg[base0];
            g1 = *(const float4*)&Wg[base0 + IDIM];
            g2 = *(const float4*)&Wg[base0 + 2 * IDIM];
            g3 = *(const float4*)&Wg[base0 + 3 * IDIM];
            u0 = *(const float4*)&Wu[base0];
            u1 = *(const float4*)&Wu[base0 + IDIM];
            u2 = *(const float4*)&Wu[base0 + 2 * IDIM];
            u3 = *(const float4*)&Wu[base0 + 3 * IDIM];
            cvt_store_quad(g0, g1, g2, g3, SB,        SB +  4096, c4A, halfA, j0A);
            cvt_store_quad(u0, u1, u2, u3, SB + 8192, SB + 12288, c4A, halfA, j0A);
            size_t b1 = base0 + 32 * (size_t)IDIM;
            g0 = *(const float4*)&Wg[b1];
            g1 = *(const float4*)&Wg[b1 + IDIM];
            g2 = *(const float4*)&Wg[b1 + 2 * IDIM];
            g3 = *(const float4*)&Wg[b1 + 3 * IDIM];
            u0 = *(const float4*)&Wu[b1];
            u1 = *(const float4*)&Wu[b1 + IDIM];
            u2 = *(const float4*)&Wu[b1 + 2 * IDIM];
            u3 = *(const float4*)&Wu[b1 + 3 * IDIM];
        }
        for (int kb = 0; kb < KB_A; ++kb) {
            __syncthreads();               // buf[kb&1] ready, buf[!(kb&1)] free
            // fragment loads first (partial dscnt wait before WMMA)
            unsigned short* BC = SB + (kb & 1) * 16384;
            v16bf ahi = *(const v16bf*)&AHi[(kb * 32 + lane) * 16];
            v16bf alo = *(const v16bf*)&ALo[(kb * 32 + lane) * 16];
            v16bf ghi = *(const v16bf*)&BC[(wave * 32 + lane) * 16];
            v16bf glo = *(const v16bf*)&BC[4096  + (wave * 32 + lane) * 16];
            v16bf uhi = *(const v16bf*)&BC[8192  + (wave * 32 + lane) * 16];
            v16bf ulo = *(const v16bf*)&BC[12288 + (wave * 32 + lane) * 16];
            if (kb + 1 < KB_A) {           // convert step kb+1 (loads issued
                                           // one iteration ago -> latency hid)
                unsigned short* BN = SB + ((kb + 1) & 1) * 16384;
                cvt_store_quad(g0, g1, g2, g3, BN,        BN +  4096, c4A, halfA, j0A);
                cvt_store_quad(u0, u1, u2, u3, BN + 8192, BN + 12288, c4A, halfA, j0A);
            }
            if (kb + 2 < KB_A) {           // issue step kb+2 loads
                size_t base = base0 + (size_t)(kb + 2) * 32 * IDIM;
                g0 = *(const float4*)&Wg[base];
                g1 = *(const float4*)&Wg[base + IDIM];
                g2 = *(const float4*)&Wg[base + 2 * IDIM];
                g3 = *(const float4*)&Wg[base + 3 * IDIM];
                u0 = *(const float4*)&Wu[base];
                u1 = *(const float4*)&Wu[base + IDIM];
                u2 = *(const float4*)&Wu[base + 2 * IDIM];
                u3 = *(const float4*)&Wu[base + 3 * IDIM];
                if (kb + 3 < KB_A) {       // pull K+3 toward L2
                    __builtin_prefetch(&Wg[base + 32 * (size_t)IDIM], 0, 0);
                    __builtin_prefetch(&Wu[base + 32 * (size_t)IDIM], 0, 0);
                }
            }
            cg = wmma_bf16(ahi, ghi, cg);
            cg = wmma_bf16(ahi, glo, cg);
            cg = wmma_bf16(alo, ghi, cg);
            cu = wmma_bf16(ahi, uhi, cu);
            cu = wmma_bf16(ahi, ulo, cu);
            cu = wmma_bf16(alo, uhi, cu);
        }
        // SiLU(g)*u -> Hm (stored as phase-B A-fragments); v_rcp_f32 sigmoid
#pragma unroll
        for (int r = 0; r < 8; ++r) {
            int m = r + ((lane >= 16) ? 8 : 0);
            int n = lane & 15;
            float g = cg[r], u = cu[r];
            float hv = g * __builtin_amdgcn_rcpf(1.0f + __expf(-g)) * u;
            int icol = colbase + wave * 16 + n;
            int kb2 = icol >> 5, kr = icol & 31;
            int lane2; int j2 = a_lane_j(m, kr, lane2);
            split_store(hv, HmHi, HmLo, (kb2 * 32 + lane2) * 16 + j2);
        }
    }

    // -------- Phase B: O = Hm @ Wd, quarter-pipelined staging -------------
    __syncthreads();                       // phase B reuses the A-tile region
    v8f co[8];
#pragma unroll
    for (int z = 0; z < 8; ++z) { v8f zz = {}; co[z] = zz; }

    // per-thread staging coords (2 items: 8 row-quads x 64 col4 per quarter)
    float4 d[2][4];
    int   rqi[2], c4i[2], hfi[2], j0i[2];
#pragma unroll
    for (int q = 0; q < 2; ++q) {
        int li = tid + q * 256;
        rqi[q] = li >> 6;
        c4i[q] = (li & 63) * 4;
        hfi[q] = ((rqi[q] * 4) >= 16) ? 16 : 0;
        j0i[q] = (rqi[q] * 4) & 15;
    }
    {   // prologue: stage step 0 into buffer 0, issue step-1 loads
#pragma unroll
        for (int q = 0; q < 2; ++q) {
            size_t base = ((size_t)e * IDIM + rqi[q] * 4) * HDIM + c4i[q];
            d[q][0] = *(const float4*)&Wd[base];
            d[q][1] = *(const float4*)&Wd[base + HDIM];
            d[q][2] = *(const float4*)&Wd[base + 2 * HDIM];
            d[q][3] = *(const float4*)&Wd[base + 3 * HDIM];
            cvt_store_quad(d[q][0], d[q][1], d[q][2], d[q][3],
                           R, R + 8192, c4i[q], hfi[q], j0i[q]);
        }
#pragma unroll
        for (int q = 0; q < 2; ++q) {      // step 1 = (kb=0, qd=1)
            size_t base = ((size_t)e * IDIM + rqi[q] * 4) * HDIM + 256 + c4i[q];
            d[q][0] = *(const float4*)&Wd[base];
            d[q][1] = *(const float4*)&Wd[base + HDIM];
            d[q][2] = *(const float4*)&Wd[base + 2 * HDIM];
            d[q][3] = *(const float4*)&Wd[base + 3 * HDIM];
        }
    }
    for (int kb = 0; kb < KB_B; ++kb) {
        v16bf ahi = *(const v16bf*)&HmHi[(kb * 32 + lane) * 16];
        v16bf alo = *(const v16bf*)&HmLo[(kb * 32 + lane) * 16];
#pragma unroll
        for (int qd = 0; qd < 4; ++qd) {   // buffer parity = qd&1 (static)
            __syncthreads();
            const int s = kb * 4 + qd;
            // fragment loads first
            unsigned short* DB = R + (qd & 1) * 16384;
            v16bf bhi0 = *(const v16bf*)&DB[((wave * 2 + 0) * 32 + lane) * 16];
            v16bf blo0 = *(const v16bf*)&DB[8192 + ((wave * 2 + 0) * 32 + lane) * 16];
            v16bf bhi1 = *(const v16bf*)&DB[((wave * 2 + 1) * 32 + lane) * 16];
            v16bf blo1 = *(const v16bf*)&DB[8192 + ((wave * 2 + 1) * 32 + lane) * 16];
            if (s + 1 < KB_B * 4) {        // convert step s+1 (loads from the
                                           // previous iteration)
                unsigned short* DN = R + ((qd + 1) & 1) * 16384;
#pragma unroll
                for (int q = 0; q < 2; ++q)
                    cvt_store_quad(d[q][0], d[q][1], d[q][2], d[q][3],
                                   DN, DN + 8192, c4i[q], hfi[q], j0i[q]);
            }
            if (s + 2 < KB_B * 4) {        // issue step s+2 loads
                const int kb2 = (s + 2) >> 2, qd2 = (s + 2) & 3;
#pragma unroll
                for (int q = 0; q < 2; ++q) {
                    size_t base = ((size_t)e * IDIM + (kb2 * 32 + rqi[q] * 4)) * HDIM
                                + qd2 * 256 + c4i[q];
                    d[q][0] = *(const float4*)&Wd[base];
                    d[q][1] = *(const float4*)&Wd[base + HDIM];
                    d[q][2] = *(const float4*)&Wd[base + 2 * HDIM];
                    d[q][3] = *(const float4*)&Wd[base + 3 * HDIM];
                    if (qd2 == 0 && kb2 + 1 < KB_B)
                        __builtin_prefetch(&Wd[base + 32 * (size_t)HDIM], 0, 0);
                }
            }
            co[qd * 2 + 0] = wmma_bf16(ahi, bhi0, co[qd * 2 + 0]);
            co[qd * 2 + 0] = wmma_bf16(ahi, blo0, co[qd * 2 + 0]);
            co[qd * 2 + 0] = wmma_bf16(alo, bhi0, co[qd * 2 + 0]);
            co[qd * 2 + 1] = wmma_bf16(ahi, bhi1, co[qd * 2 + 1]);
            co[qd * 2 + 1] = wmma_bf16(ahi, blo1, co[qd * 2 + 1]);
            co[qd * 2 + 1] = wmma_bf16(alo, bhi1, co[qd * 2 + 1]);
        }
    }

    // weighted combine: each out element gets exactly 2 commutative f32 adds
#pragma unroll
    for (int qd = 0; qd < 4; ++qd) {
#pragma unroll
        for (int cb = 0; cb < 2; ++cb) {
            int col = (qd * 16 + wave * 2 + cb) * 16 + (lane & 15);
#pragma unroll
            for (int r = 0; r < 8; ++r) {
                int m = r + ((lane >= 16) ? 8 : 0);
                int t = toks[m];
                if (t >= 0)
                    atomicAdd(&out[(size_t)t * HDIM + col],
                              tws[m] * co[qd * 2 + cb][r]);
            }
        }
    }
}

// ---------------------------------------------------------------------------
extern "C" void kernel_launch(void* const* d_in, const int* in_sizes, int n_in,
                              void* d_out, int out_size, void* d_ws, size_t ws_size,
                              hipStream_t stream) {
    (void)in_sizes; (void)n_in; (void)out_size; (void)ws_size;
    const float* x     = (const float*)d_in[0];
    const float* Wgate = (const float*)d_in[1];
    const float* bgate = (const float*)d_in[2];
    const float* Wg    = (const float*)d_in[3];
    const float* Wu    = (const float*)d_in[4];
    const float* Wd    = (const float*)d_in[5];
    float* out    = (float*)d_out;
    float* logits = out + (size_t)NTOK * HDIM;

    int*   cnt    = (int*)d_ws;
    int*   idxbuf = cnt + NEXP;
    float* wbuf   = (float*)(idxbuf + NEXP * NTOK);

    (void)hipFuncSetAttribute(reinterpret_cast<const void*>(moe_expert_kernel),
                              hipFuncAttributeMaxDynamicSharedMemorySize,
                              SMEM_BYTES);

    moe_init_kernel<<<(NEXP * NTOK + 255) / 256, 256, 0, stream>>>(cnt, idxbuf);
    moe_zero_out_kernel<<<(NTOK * HDIM + 255) / 256, 256, 0, stream>>>(out);
    moe_gate_kernel<<<(NTOK * 32 + 255) / 256, 256, 0, stream>>>(
        x, Wgate, bgate, logits, cnt, idxbuf, wbuf);
    moe_expert_kernel<<<dim3(NTOK / TW, NEXP), 256, SMEM_BYTES, stream>>>(
        x, Wg, Wu, Wd, cnt, idxbuf, wbuf, out);
}
